// Contrastive_Loss_32409823215842
// MI455X (gfx1250) — matrix operand order repaired and checked by reading, not compiled, not run on hardware
//
#include <hip/hip_runtime.h>

typedef __attribute__((ext_vector_type(16))) _Float16 v16h;
typedef __attribute__((ext_vector_type(8)))  float    v8f;
typedef __attribute__((ext_vector_type(4)))  float    v4f;

#define BB   4096
#define NN   8192
#define DD   256
#define NBAND  32            // bands of 16 row tiles (256 rows)
#define NSPLIT 16            // column splits per band
#define LDSPITCH 264         // padded halves per staged row (bank-conflict-free)

// ---------------------------------------------------------------------------
// Kernel 1: row-normalize -> f16 matrix; emit D[row] = 2*||zn_f16||^2
// (f16-quantized sim diagonal); zero the SCOL atomic accumulator.
// ---------------------------------------------------------------------------
__global__ __launch_bounds__(256) void nrm_kernel(const float* __restrict__ zi,
                                                  const float* __restrict__ zj,
                                                  _Float16* __restrict__ znh,
                                                  float* __restrict__ D,
                                                  float* __restrict__ SCOL) {
    const int lane = threadIdx.x & 31;
    const int wv   = threadIdx.x >> 5;
    const int row  = blockIdx.x * 8 + wv;
    const float* src = (row < BB) ? (zi + (size_t)row * DD)
                                  : (zj + (size_t)(row - BB) * DD);
    const v4f* p4 = (const v4f*)src;
    v4f a = p4[lane * 2 + 0];
    v4f b = p4[lane * 2 + 1];
    float ss = a.x*a.x + a.y*a.y + a.z*a.z + a.w*a.w
             + b.x*b.x + b.y*b.y + b.z*b.z + b.w*b.w;
    #pragma unroll
    for (int m = 1; m < 32; m <<= 1) ss += __shfl_xor(ss, m, 32);
    const float inv = 1.0f / fmaxf(sqrtf(ss), 1e-8f);

    union { _Float16 h[8]; v4f q; } u;
    u.h[0] = (_Float16)(a.x * inv);  u.h[1] = (_Float16)(a.y * inv);
    u.h[2] = (_Float16)(a.z * inv);  u.h[3] = (_Float16)(a.w * inv);
    u.h[4] = (_Float16)(b.x * inv);  u.h[5] = (_Float16)(b.y * inv);
    u.h[6] = (_Float16)(b.z * inv);  u.h[7] = (_Float16)(b.w * inv);
    *(v4f*)(znh + (size_t)row * DD + lane * 8) = u.q;

    float sd = 0.0f;
    #pragma unroll
    for (int t = 0; t < 8; ++t) { float x = (float)u.h[t]; sd += x * x; }
    #pragma unroll
    for (int m = 1; m < 32; m <<= 1) sd += __shfl_xor(sd, m, 32);
    if (lane == 0) { D[row] = 2.0f * sd; SCOL[row] = 0.0f; }
}

// ---------------------------------------------------------------------------
// Kernel 2: POS[row] = 2 * <zn_row, zn_{row^4096}>.
// ---------------------------------------------------------------------------
__global__ __launch_bounds__(256) void pos_kernel(const _Float16* __restrict__ znh,
                                                  float* __restrict__ POS) {
    const int lane = threadIdx.x & 31;
    const int wv   = threadIdx.x >> 5;
    const int row  = blockIdx.x * 8 + wv;
    const int par  = row ^ BB;
    union { _Float16 h[8]; v4f q; } ua, ub;
    ua.q = *(const v4f*)(znh + (size_t)row * DD + lane * 8);
    ub.q = *(const v4f*)(znh + (size_t)par * DD + lane * 8);
    float s = 0.0f;
    #pragma unroll
    for (int t = 0; t < 8; ++t) s = fmaf((float)ua.h[t], (float)ub.h[t], s);
    #pragma unroll
    for (int m = 1; m < 32; m <<= 1) s += __shfl_xor(s, m, 32);
    if (lane == 0) POS[row] = 2.0f * s;
}

// ---------------------------------------------------------------------------
// Kernel 3: lower-triangle of sim via WMMA, with GLOBAL_LOAD_ASYNC_TO_LDS
// double-buffered column-tile staging (ASYNCcnt-tracked DMA, no VGPR trip).
// Grid (NSPLIT, NBAND). Band n = 31 - blockIdx.y owns row tiles [16n,16n+16);
// wave w owns rtA=16n+2w, rtB=rtA+1 (A register-resident, R=2 reuse).
// launch_bounds(256,1): relax pressure heuristics so all 8 B fragments stay
// live and the DS loads batch with progressive dscnt waits.
// ---------------------------------------------------------------------------
__global__ __launch_bounds__(256, 1) void sim_kernel(const _Float16* __restrict__ znh,
                                                     float* __restrict__ SP,    // [NSPLIT][NN]
                                                     float* __restrict__ SCOL)  // [NN]
{
    __shared__ _Float16 lds[2][16 * LDSPITCH];   // 2 x 8448 B

    const int lane = threadIdx.x & 31;
    const int w    = threadIdx.x >> 5;
    const int s    = blockIdx.x;
    const int n    = (NBAND - 1) - blockIdx.y;   // biggest bands first
    const int half = lane >> 4;
    const int l16  = lane & 15;

    const int rtA = 16 * n + 2 * w;
    const int rtB = rtA + 1;
    const int tiles = n + 1;
    const int ct0 = s * tiles;

    // ---- A fragments for both row tiles (ISA 16-bit A layout) ----
    v16h A0[8], A1[8];
    {
        const _Float16* pa0 = znh + (size_t)((rtA << 4) + l16) * DD + half * 8;
        const _Float16* pa1 = znh + (size_t)((rtB << 4) + l16) * DD + half * 8;
        #pragma unroll
        for (int k = 0; k < 8; ++k) {
            union { v16h h; v4f q[2]; } u;
            u.q[0] = *(const v4f*)(pa0 + k * 32);
            u.q[1] = *(const v4f*)(pa0 + k * 32 + 16);
            A0[k] = u.h;
            u.q[0] = *(const v4f*)(pa1 + k * 32);
            u.q[1] = *(const v4f*)(pa1 + k * 32 + 16);
            A1[k] = u.h;
        }
    }

    float racc0[8], racc1[8];
    #pragma unroll
    for (int v = 0; v < 8; ++v) { racc0[v] = 0.0f; racc1[v] = 0.0f; }

    // staging: thread -> (row sr, 32B chunk sc) of the 16x256 column tile
    const int sr = threadIdx.x >> 4;
    const int sc = threadIdx.x & 15;
    // raw LDS byte offsets (low 32 bits of the LDS aperture address)
    const unsigned lo0 = (unsigned)(size_t)&lds[0][sr * LDSPITCH + sc * 16];
    const unsigned lo1 = (unsigned)(size_t)&lds[1][sr * LDSPITCH + sc * 16];

    // stage first tile: async DMA global->LDS (two b128, offset applies to both sides)
    {
        const _Float16* g = znh + (size_t)((ct0 << 4) + sr) * DD + sc * 16;
        asm volatile("global_load_async_to_lds_b128 %0, %1, off\n\t"
                     "global_load_async_to_lds_b128 %0, %1, off offset:16"
                     :: "v"(lo0), "v"(g) : "memory");
    }
    asm volatile("s_wait_asynccnt 0x0" ::: "memory");
    __syncthreads();

    for (int i = 0; i < tiles; ++i) {
        const int ct = ct0 + i;
        // async-stage next tile into the other buffer
        if (i + 1 < tiles) {
            const _Float16* g = znh + (size_t)(((ct + 1) << 4) + sr) * DD + sc * 16;
            const unsigned dst = ((i + 1) & 1) ? lo1 : lo0;
            asm volatile("global_load_async_to_lds_b128 %0, %1, off\n\t"
                         "global_load_async_to_lds_b128 %0, %1, off offset:16"
                         :: "v"(dst), "v"(g) : "memory");
        }

        if (ct <= rtB) {                      // wave-uniform: inside triangle
            const _Float16* lp = &lds[i & 1][l16 * LDSPITCH + half * 8];

            // ---- preload all 8 B fragments (16 ds_load_b128 in flight) ----
            v16h Bf[8];
            #pragma unroll
            for (int k = 0; k < 8; ++k) {
                union { v16h h; v4f q[2]; } u;
                u.q[0] = *(const v4f*)(lp + k * 32);
                u.q[1] = *(const v4f*)(lp + k * 32 + 16);
                Bf[k] = u.h;
            }

            v8f C0 = {}, C1 = {};
            #pragma unroll
            for (int k = 0; k < 8; ++k) {
                C0 = __builtin_amdgcn_wmma_f32_16x16x32_f16(
                        false, A0[k], false, Bf[k], (short)0, C0, false, false);
                C1 = __builtin_amdgcn_wmma_f32_16x16x32_f16(
                        false, A1[k], false, Bf[k], (short)0, C1, false, false);
            }

            const bool doRow0 = (ct <= rtA);  // false only for ct == rtB tile
            const bool doCol0 = (ct <  rtA);
            const bool doCol1 = (ct <  rtB);
            float col = 0.0f;
            #pragma unroll
            for (int v = 0; v < 8; ++v) {
                float e0 = __expf(fmaf(C0[v], 2.0f, -2.0f));
                float e1 = __expf(fmaf(C1[v], 2.0f, -2.0f));
                if (doRow0) racc0[v] += e0;
                racc1[v] += e1;
                if (doCol0) col += e0;
                if (doCol1) col += e1;
            }
            if (doCol1 || doCol0) {           // symmetric column contribution
                col += __shfl_xor(col, 16, 32);
                if (lane < 16)
                    atomicAdd(&SCOL[(ct << 4) + l16], col);
            }
        }

        // async LDS writes must land before the barrier releases readers
        asm volatile("s_wait_asynccnt 0x0" ::: "memory");
        __syncthreads();
    }

    // ---- per-row reduction over the 16 column lanes; deterministic store ----
    #pragma unroll
    for (int v = 0; v < 8; ++v) {
        float x0 = racc0[v], x1 = racc1[v];
        x0 += __shfl_xor(x0, 1, 32); x1 += __shfl_xor(x1, 1, 32);
        x0 += __shfl_xor(x0, 2, 32); x1 += __shfl_xor(x1, 2, 32);
        x0 += __shfl_xor(x0, 4, 32); x1 += __shfl_xor(x1, 4, 32);
        x0 += __shfl_xor(x0, 8, 32); x1 += __shfl_xor(x1, 8, 32);
        if (l16 == v) {
            SP[(size_t)s * NN + (rtA << 4) + half * 8 + v] = x0;
            SP[(size_t)s * NN + (rtB << 4) + half * 8 + v] = x1;
        }
    }
}

// ---------------------------------------------------------------------------
// Kernel 4: loss = mean_i( 2 + log(Σ_p SP[p][i] + SCOL[i] - exp(D[i]-2)) - POS[i] )
// ---------------------------------------------------------------------------
__global__ __launch_bounds__(256) void fin_kernel(const float* __restrict__ SP,
                                                  const float* __restrict__ SCOL,
                                                  const float* __restrict__ D,
                                                  const float* __restrict__ POS,
                                                  float* __restrict__ out) {
    __shared__ float sm[256];
    float t = 0.0f;
    for (int i = threadIdx.x; i < NN; i += 256) {
        float sv = SCOL[i] - __expf(D[i] - 2.0f);
        #pragma unroll
        for (int p = 0; p < NSPLIT; ++p) sv += SP[(size_t)p * NN + i];
        t += 2.0f + logf(sv) - POS[i];
    }
    sm[threadIdx.x] = t;
    __syncthreads();
    #pragma unroll
    for (int off = 128; off > 0; off >>= 1) {
        if (threadIdx.x < off) sm[threadIdx.x] += sm[threadIdx.x + off];
        __syncthreads();
    }
    if (threadIdx.x == 0) out[0] = sm[0] / (float)NN;
}

// ---------------------------------------------------------------------------
extern "C" void kernel_launch(void* const* d_in, const int* in_sizes, int n_in,
                              void* d_out, int out_size, void* d_ws, size_t ws_size,
                              hipStream_t stream) {
    const float* zi = (const float*)d_in[0];
    const float* zj = (const float*)d_in[1];

    char* ws = (char*)d_ws;
    _Float16* znh = (_Float16*)ws;   ws += (size_t)NN * DD * sizeof(_Float16);  // 4 MB
    float* SP   = (float*)ws;        ws += (size_t)NSPLIT * NN * sizeof(float); // 512 KB
    float* SCOL = (float*)ws;        ws += (size_t)NN * sizeof(float);
    float* D    = (float*)ws;        ws += (size_t)NN * sizeof(float);
    float* POS  = (float*)ws;
    float* out  = (float*)d_out;

    nrm_kernel<<<NN / 8, 256, 0, stream>>>(zi, zj, znh, D, SCOL);
    pos_kernel<<<NN / 8, 256, 0, stream>>>(znh, POS);
    sim_kernel<<<dim3(NSPLIT, NBAND), 256, 0, stream>>>(znh, SP, SCOL);
    fin_kernel<<<1, 256, 0, stream>>>(SP, SCOL, D, POS, out);
}
